// TinyLoRALinear_47571057771012
// MI455X (gfx1250) — matrix operand order repaired and checked by reading, not compiled, not run on hardware
//
#include <hip/hip_runtime.h>

// ---------------------------------------------------------------------------
// TinyLoRA linear: out = x @ (W_base + U diag(S*(v@P)) Vh)^T + bias
// Fold the rank-2 LoRA term into the weight tile while staging it to LDS, then
// run a single bf16x3 (hi/lo split) WMMA GEMM with f32 accumulation.
// Block tile 128x128, 512 threads (16 wave32s, 4x4 grid, 32x32 per wave),
// K-step 32, ping-pong double-buffered LDS (one barrier per K-step).
// Per-wave VGPR budget kept < 256 to avoid s_set_vgpr_msb mode switching.
// ---------------------------------------------------------------------------

typedef __attribute__((ext_vector_type(16))) __bf16 v16bf;
typedef __attribute__((ext_vector_type(8)))  float  v8f;

#define IN_DIM  4096
#define OUT_DIM 4096
#define BM 128
#define BN 128
#define BK 32
#define LDK 40   // padded LDS row length (bf16 elems) -> 80B stride, bank spread

#define WMMA_BF16(A, B, C) \
  __builtin_amdgcn_wmma_f32_16x16x32_bf16(false, (A), false, (B), (short)0, (C), false, false)

// A-operand fragment: 16x32 tile (row-major in LDS). Lane l (0-15): row=l,
// K = 0..7 (VGPR0-3) and 16..23 (VGPR4-7); lane l+16: row=l, K = 8..15, 24..31.
__device__ __forceinline__ v16bf ld_fragA(const __bf16* tile, int lane) {
  const int row  = lane & 15;
  const int half = lane >> 4;
  const __bf16* p = tile + row * LDK + half * 8;
  union { uint4 q[2]; v16bf v; } u;
  u.q[0] = *(const uint4*)(p);        // K chunk [half*8, half*8+8)
  u.q[1] = *(const uint4*)(p + 16);   // K chunk [16+half*8, ...)
  return u.v;
}

// B-operand fragment: B is K x N; we hold W as N x K rows in LDS, so column n
// of B is W-row n. Lanes 0-15: K=0..7, 16..23; lanes 16-31: K=8..15, 24..31.
__device__ __forceinline__ v16bf ld_fragB(const __bf16* tile, int lane) {
  const int col  = lane & 15;
  const int half = lane >> 4;
  const __bf16* p = tile + col * LDK + half * 8;
  union { uint4 q[2]; v16bf v; } u;
  u.q[0] = *(const uint4*)(p);
  u.q[1] = *(const uint4*)(p + 16);
  return u.v;
}

// Split an fp32 quad into bf16 hi + bf16 residual lo, store both as 8B packets.
__device__ __forceinline__ void split4(float4 w, __bf16* hi, __bf16* lo, int idx) {
  union { __bf16 h[4]; uint2 u; } ph, pl;
  const float f0 = w.x, f1 = w.y, f2 = w.z, f3 = w.w;
  const __bf16 h0 = (__bf16)f0, h1 = (__bf16)f1, h2 = (__bf16)f2, h3 = (__bf16)f3;
  ph.h[0] = h0; ph.h[1] = h1; ph.h[2] = h2; ph.h[3] = h3;
  pl.h[0] = (__bf16)(f0 - (float)h0);
  pl.h[1] = (__bf16)(f1 - (float)h1);
  pl.h[2] = (__bf16)(f2 - (float)h2);
  pl.h[3] = (__bf16)(f3 - (float)h3);
  *(uint2*)(hi + idx) = ph.u;
  *(uint2*)(lo + idx) = pl.u;
}

// One K-step: consume LDS buffer CUR with 12 WMMAs while prefetching +
// LoRA-folding + bf16-splitting the next K-tile into buffer NXT.
#define K_STEP(KCUR, CUR, NXT)                                                 \
  {                                                                            \
    const int knext = (KCUR) + BK;                                             \
    const bool do_next = knext < IN_DIM;                                       \
    float4 xr[2], wr[2];                                                       \
    float4 vh0 = {}, vh1 = {};                                                 \
    if (do_next) {                                                             \
      vh0 = *(const float4*)(Vh + knext + lk);                                 \
      vh1 = *(const float4*)(Vh + IN_DIM + knext + lk);                        \
      _Pragma("unroll")                                                        \
      for (int i = 0; i < 2; ++i) {                                            \
        const int r = lr + i * 64;                                             \
        xr[i] = *(const float4*)(x  + (size_t)(mblock + r) * IN_DIM + knext + lk); \
        wr[i] = *(const float4*)(Wb + (size_t)(nblock + r) * IN_DIM + knext + lk); \
      }                                                                        \
    }                                                                          \
    v16bf axh[2], axl[2], bwh[2], bwl[2];                                      \
    _Pragma("unroll")                                                          \
    for (int mt = 0; mt < 2; ++mt) {                                           \
      const int rbase = (wm * 32 + mt * 16) * LDK;                             \
      axh[mt] = ld_fragA(&sXhi[CUR][rbase], lane);                             \
      axl[mt] = ld_fragA(&sXlo[CUR][rbase], lane);                             \
    }                                                                          \
    _Pragma("unroll")                                                          \
    for (int nt = 0; nt < 2; ++nt) {                                           \
      const int rbase = (wn * 32 + nt * 16) * LDK;                             \
      bwh[nt] = ld_fragB(&sWhi[CUR][rbase], lane);                             \
      bwl[nt] = ld_fragB(&sWlo[CUR][rbase], lane);                             \
    }                                                                          \
    if (do_next) {                                                             \
      _Pragma("unroll")                                                        \
      for (int i = 0; i < 2; ++i) {                                            \
        const int r = lr + i * 64;                                             \
        split4(xr[i], sXhi[NXT], sXlo[NXT], r * LDK + lk);                     \
        float4 w = wr[i];                                                      \
        w.x += u0s[i] * vh0.x + u1s[i] * vh1.x;                                \
        w.y += u0s[i] * vh0.y + u1s[i] * vh1.y;                                \
        w.z += u0s[i] * vh0.z + u1s[i] * vh1.z;                                \
        w.w += u0s[i] * vh0.w + u1s[i] * vh1.w;                                \
        split4(w, sWhi[NXT], sWlo[NXT], r * LDK + lk);                         \
      }                                                                        \
    }                                                                          \
    _Pragma("unroll")                                                          \
    for (int mt = 0; mt < 2; ++mt) {                                           \
      _Pragma("unroll")                                                        \
      for (int nt = 0; nt < 2; ++nt) {                                         \
        acc[mt][nt] = WMMA_BF16(axh[mt], bwh[nt], acc[mt][nt]);                \
        acc[mt][nt] = WMMA_BF16(axh[mt], bwl[nt], acc[mt][nt]);                \
        acc[mt][nt] = WMMA_BF16(axl[mt], bwh[nt], acc[mt][nt]);                \
      }                                                                        \
    }                                                                          \
    __syncthreads();                                                           \
  }

__global__ __launch_bounds__(512, 1)
void tinylora_wmma_kernel(const float* __restrict__ x,
                          const float* __restrict__ Wb,
                          const float* __restrict__ U,     // [OUT, 2]
                          const float* __restrict__ S,     // [2]
                          const float* __restrict__ Vh,    // [2, IN]
                          const float* __restrict__ P,     // [64, 2]
                          const float* __restrict__ v,     // [64]
                          const float* __restrict__ bias,  // [OUT]
                          float* __restrict__ out) {
  // Ping-pong double buffers: 2 * 4 * 128*40*2B = 80 KB (WGP has 320 KB).
  __shared__ __align__(16) __bf16 sXhi[2][BM * LDK];
  __shared__ __align__(16) __bf16 sXlo[2][BM * LDK];
  __shared__ __align__(16) __bf16 sWhi[2][BN * LDK];
  __shared__ __align__(16) __bf16 sWlo[2][BN * LDK];

  const int tid  = threadIdx.x;
  const int lane = tid & 31;
  const int wave = tid >> 5;   // 0..15
  const int wm   = wave & 3;   // 32-row quarter of the block tile
  const int wn   = wave >> 2;  // 32-col quarter

  const int mblock = blockIdx.y * BM;
  const int nblock = blockIdx.x * BN;

  // s_vp[r] = S[r] * sum_u v[u] * P[u][r]   (tiny, uniform across the block)
  float a0 = 0.f, a1 = 0.f;
  #pragma unroll 8
  for (int u = 0; u < 64; ++u) {
    const float vu = v[u];
    a0 += vu * P[2 * u + 0];
    a1 += vu * P[2 * u + 1];
  }
  const float s0 = S[0] * a0;
  const float s1 = S[1] * a1;

  // Staging decomposition: each thread loads 2 rows (stride 64) x 4 K values.
  const int lr = tid >> 3;         // 0..63 base row
  const int lk = (tid & 7) << 2;   // 0,4,...,28

  // Per-thread LoRA row coefficients (rows fixed across the whole K loop).
  float u0s[2], u1s[2];
  #pragma unroll
  for (int i = 0; i < 2; ++i) {
    const int n = nblock + lr + i * 64;
    u0s[i] = U[2 * n + 0] * s0;
    u1s[i] = U[2 * n + 1] * s1;
  }

  v8f acc[2][2];
  {
    const v8f z = {};
    #pragma unroll
    for (int mt = 0; mt < 2; ++mt)
      #pragma unroll
      for (int nt = 0; nt < 2; ++nt)
        acc[mt][nt] = z;
  }

  // ---- prologue: stage K-tile 0 into buffer 0 ----------------------------
  {
    const float4 vh0 = *(const float4*)(Vh + lk);
    const float4 vh1 = *(const float4*)(Vh + IN_DIM + lk);
    #pragma unroll
    for (int i = 0; i < 2; ++i) {
      const int r = lr + i * 64;
      const float4 xv = *(const float4*)(x + (size_t)(mblock + r) * IN_DIM + lk);
      split4(xv, sXhi[0], sXlo[0], r * LDK + lk);
      float4 w = *(const float4*)(Wb + (size_t)(nblock + r) * IN_DIM + lk);
      w.x += u0s[i] * vh0.x + u1s[i] * vh1.x;
      w.y += u0s[i] * vh0.y + u1s[i] * vh1.y;
      w.z += u0s[i] * vh0.z + u1s[i] * vh1.z;
      w.w += u0s[i] * vh0.w + u1s[i] * vh1.w;
      split4(w, sWhi[0], sWlo[0], r * LDK + lk);
    }
  }
  __syncthreads();

  // ---- main loop: unrolled x2 so ping-pong buffer indices are literals ----
  for (int k0 = 0; k0 < IN_DIM; k0 += 2 * BK) {
    K_STEP(k0, 0, 1)
    K_STEP(k0 + BK, 1, 0)
  }

  // ---- epilogue: bias + store (C layout: VGPR j -> M=j/j+8, lane -> N) ---
  #pragma unroll
  for (int nt = 0; nt < 2; ++nt) {
    const int col  = nblock + wn * 32 + nt * 16 + (lane & 15);
    const float bv = bias[col];
    #pragma unroll
    for (int mt = 0; mt < 2; ++mt) {
      const int rowbase = mblock + wm * 32 + mt * 16 + (lane >> 4) * 8;
      #pragma unroll
      for (int j = 0; j < 8; ++j) {
        out[(size_t)(rowbase + j) * OUT_DIM + col] = acc[mt][nt][j] + bv;
      }
    }
  }
}

extern "C" void kernel_launch(void* const* d_in, const int* in_sizes, int n_in,
                              void* d_out, int out_size, void* d_ws, size_t ws_size,
                              hipStream_t stream) {
  (void)n_in; (void)out_size; (void)d_ws; (void)ws_size;
  const float* x    = (const float*)d_in[0];
  const float* Wb   = (const float*)d_in[1];
  const float* U    = (const float*)d_in[2];
  const float* S    = (const float*)d_in[3];
  const float* Vh   = (const float*)d_in[4];
  const float* P    = (const float*)d_in[5];
  const float* v    = (const float*)d_in[6];
  const float* bias = (const float*)d_in[7];
  float* out = (float*)d_out;

  const int Mrows = in_sizes[0] / IN_DIM;  // B*T = 8192
  dim3 grid(OUT_DIM / BN, Mrows / BM);
  dim3 block(512);
  tinylora_wmma_kernel<<<grid, block, 0, stream>>>(x, Wb, U, S, Vh, P, v, bias, out);
}
